// TabMixer_7584912244830
// MI455X (gfx1250) — matrix operand order repaired and checked by reading, not compiled, not run on hardware
//
#include <hip/hip_runtime.h>
#include <math.h>

typedef unsigned short u16;
typedef unsigned int   u32;
typedef unsigned long long u64;

typedef __attribute__((ext_vector_type(16))) __bf16 v16bf;
typedef __attribute__((ext_vector_type(8)))  float  v8f;

// ---------------- bf16 helpers (raw-bit storage as u16) ----------------
__device__ __forceinline__ u16 f2bf(float f) {
    union { float f; u32 u; } v; v.f = f;
    u32 r = v.u + 0x7fffu + ((v.u >> 16) & 1u);   // round-to-nearest-even
    return (u16)(r >> 16);
}
__device__ __forceinline__ float bf2f(u16 h) {
    union { u32 u; float f; } v; v.u = ((u32)h) << 16;
    return v.f;
}

// -------- CDNA5 async global->LDS staging (ASYNCcnt-tracked, ISA ch.8/15.18) --------
__device__ __forceinline__ void async_load_b128(u32 lds_off, const void* gaddr) {
    asm volatile("global_load_async_to_lds_b128 %0, %1, off"
                 :: "v"(lds_off), "v"((u64)(size_t)gaddr) : "memory");
}
__device__ __forceinline__ void wait_async0() {
    asm volatile("s_wait_asynccnt 0" ::: "memory");
}

// ---------------- tiling ----------------
#define BM 64
#define BN 256
#define BK 32
#define APITCH 40     // u16 units; 80B rows -> 16B-aligned async B128 dst
#define BPITCH 264    // u16 units; 528B rows -> 16B-aligned async B128 dst

#define EP_IF       0   // spike = (x>=1)?1:0          -> bf16
#define EP_IF_GELU  1   // gelu(spike): {0, 0.8413447} -> bf16
#define EP_IF_SIG   2   // sigmoid(spike): {0.5, 0.7310586} -> bf16
#define EP_SIG_F32  3   // sigmoid(x) -> f32
#define EP_LIN      4   // identity -> bf16
#define EP_RESID    5   // h = m * spike + h (in place, bf16)

// ================= WMMA bf16 GEMM  C = epilogue(A[M,K] @ B[K,N] + bias) =================
// AMODE 0: A = bf16 row-major (async-staged).  AMODE 1: A generated on the fly:
//   A[m,k] = bf16( (d[b,2r] + d[b,2r+1]*grid_eps[m*K+k]) * coeff[b] ),  b=m>>9, r=m&511
// 256 threads = 8 waves; wave grid 2(M) x 4(N); wave tile 32x64 -> 2x4 frags, 8 WMMA/K-step.
template <int AMODE, int EP>
__global__ __launch_bounds__(256)
void gemm_bf16_wmma(const u16* __restrict__ A, const u16* __restrict__ B,
                    const float* __restrict__ bias, void* __restrict__ Cout,
                    int M, int N, int K,
                    const float* __restrict__ dbuf, const float* __restrict__ coeff,
                    const float* __restrict__ geps,
                    const u16* __restrict__ mbuf, u16* __restrict__ hbuf)
{
    __shared__ u16 As[BM][APITCH];     // row-major, K contiguous
    __shared__ u16 Bs[BK][BPITCH];     // row-major, N contiguous

    const int tid  = threadIdx.x;
    const int lane = tid & 31;
    const int hi   = lane >> 4;
    const int l15  = lane & 15;
    const int wave = tid >> 5;
    const int wm   = (wave >> 2) * 32;   // 0,32
    const int wn   = (wave & 3) * 64;    // 0..192
    const int mbase = blockIdx.y * BM;
    const int nbase = blockIdx.x * BN;

    const u32 asBase = (u32)(size_t)&As[0][0];
    const u32 bsBase = (u32)(size_t)&Bs[0][0];

    v8f acc[2][4] = {};

    for (int k0 = 0; k0 < K; k0 += BK) {
        // ---- stage A tile 64x32 ----
        if (AMODE == 0) {
            int arow = tid >> 2;                 // 0..63
            int acol = (tid & 3) * 8;            // 0,8,16,24
            async_load_b128(asBase + (u32)(arow * APITCH + acol) * 2,
                            A + (size_t)(mbase + arow) * K + (k0 + acol));
        } else {
            int e = tid * 8;
            int row = e >> 5, col = e & 31;
            int mg = mbase + row;
            int b = mg >> 9, r = mg & 511;
            float mean = dbuf[b * 1024 + 2 * r];
            float sd   = dbuf[b * 1024 + 2 * r + 1];
            float cf   = coeff[b];
            const float* gp = geps + (size_t)mg * K + (k0 + col);
#pragma unroll
            for (int j = 0; j < 8; ++j)
                As[row][col + j] = f2bf((mean + sd * gp[j]) * cf);
        }
        // ---- stage B tile 32x256 row-major: 4 async B128 chunks/thread ----
#pragma unroll
        for (int c = 0; c < 4; ++c) {
            int id = tid * 4 + c;                // 0..1023
            int kr = id >> 5;                    // 0..31
            int nc = (id & 31) * 8;              // 0..248
            const u16* src = B + (size_t)(k0 + kr) * N + (nbase + nc);
            async_load_b128(bsBase + (u32)(kr * BPITCH + nc) * 2, src);
            if (c == 0 && k0 + BK < K) __builtin_prefetch(src + (size_t)BK * N, 0, 1);
        }
        wait_async0();
        __syncthreads();

        // ---- fragment gathers (documented 16-bit A/B WMMA layouts) ----
        union Frag { v16bf v; u32 u[8]; };
        Frag af[2], bfm[4];
#pragma unroll
        for (int i = 0; i < 2; ++i) {
            int m = wm + i * 16 + l15;
#pragma unroll
            for (int j = 0; j < 8; ++j) {
                int kp = ((j < 4) ? (2 * j) : (2 * j + 8)) + 8 * hi;   // A: 0-7/16-23 | 8-15/24-31
                af[i].u[j] = *(const u32*)&As[m][kp];
            }
        }
#pragma unroll
        for (int j2 = 0; j2 < 4; ++j2) {
            int n = wn + j2 * 16 + l15;
#pragma unroll
            for (int j = 0; j < 8; ++j) {
                int k = ((j < 4) ? (2 * j) : (2 * j + 8)) + 8 * hi;     // B: 0-7/16-23 | 8-15/24-31
                bfm[j2].u[j] = (u32)Bs[k][n] | ((u32)Bs[k + 1][n] << 16);
            }
        }
#pragma unroll
        for (int i = 0; i < 2; ++i)
#pragma unroll
            for (int j2 = 0; j2 < 4; ++j2)
                acc[i][j2] = __builtin_amdgcn_wmma_f32_16x16x32_bf16(
                    false, af[i].v, false, bfm[j2].v, (short)0, acc[i][j2], false, false);
        __syncthreads();
    }

    // ---- epilogue: C layout VGPR r -> M = r + 8*hi, N = lane&15 ----
#pragma unroll
    for (int i = 0; i < 2; ++i) {
#pragma unroll
        for (int j2 = 0; j2 < 4; ++j2) {
#pragma unroll
            for (int r = 0; r < 8; ++r) {
                int row = mbase + wm + i * 16 + r + 8 * hi;
                int col = nbase + wn + j2 * 16 + l15;
                float val = acc[i][j2][r] + bias[col];
                size_t idx = (size_t)row * N + col;
                if (EP == EP_IF) {
                    ((u16*)Cout)[idx] = f2bf(val >= 1.f ? 1.f : 0.f);
                } else if (EP == EP_IF_GELU) {
                    ((u16*)Cout)[idx] = f2bf(val >= 1.f ? 0.84134474f : 0.f);
                } else if (EP == EP_IF_SIG) {
                    ((u16*)Cout)[idx] = f2bf(val >= 1.f ? 0.73105858f : 0.5f);
                } else if (EP == EP_SIG_F32) {
                    ((float*)Cout)[idx] = 1.f / (1.f + expf(-val));
                } else if (EP == EP_LIN) {
                    ((u16*)Cout)[idx] = f2bf(val);
                } else { // EP_RESID : h = m * spike + h   (N==512 geometry)
                    float sp = (val >= 1.f) ? 1.f : 0.f;
                    float hv = bf2f(hbuf[idx]);
                    float mv = bf2f(mbuf[idx]);
                    hbuf[idx] = f2bf(mv * sp + hv);
                }
            }
        }
    }
}

// ================= LayerNorm over last dim (512), bf16 in/out =================
__global__ __launch_bounds__(256)
void ln_kernel(const u16* __restrict__ h, const float* __restrict__ g,
               const float* __restrict__ bb, u16* __restrict__ xn)
{
    __shared__ float red[256];
    int row = blockIdx.x, t = threadIdx.x;
    const u16* hr = h + (size_t)row * 512;
    float v0 = bf2f(hr[t]), v1 = bf2f(hr[t + 256]);
    red[t] = v0 + v1; __syncthreads();
    for (int o = 128; o > 0; o >>= 1) { if (t < o) red[t] += red[t + o]; __syncthreads(); }
    float mu = red[0] * (1.f / 512.f); __syncthreads();
    float d0 = v0 - mu, d1 = v1 - mu;
    red[t] = d0 * d0 + d1 * d1; __syncthreads();
    for (int o = 128; o > 0; o >>= 1) { if (t < o) red[t] += red[t + o]; __syncthreads(); }
    float rs = rsqrtf(red[0] * (1.f / 512.f) + 1e-5f);
    u16* xr = xn + (size_t)row * 512;
    xr[t]       = f2bf(d0 * rs * g[t] + bb[t]);
    xr[t + 256] = f2bf(d1 * rs * g[t + 256] + bb[t + 256]);
}

// ============ final LN + mean over tokens: pooled[b,c] += LN(h[b,n,:])[c]/512 ============
__global__ __launch_bounds__(256)
void final_pool_kernel(const u16* __restrict__ h, const float* __restrict__ g,
                       const float* __restrict__ bb, float* __restrict__ pooled)
{
    __shared__ float red[256];
    int row = blockIdx.x, t = threadIdx.x;
    const u16* hr = h + (size_t)row * 512;
    float v0 = bf2f(hr[t]), v1 = bf2f(hr[t + 256]);
    red[t] = v0 + v1; __syncthreads();
    for (int o = 128; o > 0; o >>= 1) { if (t < o) red[t] += red[t + o]; __syncthreads(); }
    float mu = red[0] * (1.f / 512.f); __syncthreads();
    float d0 = v0 - mu, d1 = v1 - mu;
    red[t] = d0 * d0 + d1 * d1; __syncthreads();
    for (int o = 128; o > 0; o >>= 1) { if (t < o) red[t] += red[t + o]; __syncthreads(); }
    float rs = rsqrtf(red[0] * (1.f / 512.f) + 1e-5f);
    int b = row >> 9;
    atomicAdd(&pooled[b * 512 + t],       (d0 * rs * g[t] + bb[t]) * (1.f / 512.f));
    atomicAdd(&pooled[b * 512 + t + 256], (d1 * rs * g[t + 256] + bb[t + 256]) * (1.f / 512.f));
}

// ============ head GEMM [128,512]@[512,1000] + softmax ============
__global__ __launch_bounds__(256)
void head_softmax_kernel(const float* __restrict__ pooled, const float* __restrict__ W,
                         const float* __restrict__ bias, float* __restrict__ out)
{
    __shared__ float xs[512];
    __shared__ float logits[1000];
    __shared__ float red[256];
    int b = blockIdx.x, t = threadIdx.x;
    xs[t] = pooled[b * 512 + t]; xs[t + 256] = pooled[b * 512 + t + 256];
    __syncthreads();
    float lmax = -1e30f;
    for (int c = t; c < 1000; c += 256) {
        float acc = bias[c];
        for (int k = 0; k < 512; ++k) acc += xs[k] * W[(size_t)k * 1000 + c];
        logits[c] = acc;
        lmax = fmaxf(lmax, acc);
    }
    red[t] = lmax; __syncthreads();
    for (int o = 128; o > 0; o >>= 1) { if (t < o) red[t] = fmaxf(red[t], red[t + o]); __syncthreads(); }
    float mx = red[0]; __syncthreads();
    float lsum = 0.f;
    for (int c = t; c < 1000; c += 256) { float e = expf(logits[c] - mx); logits[c] = e; lsum += e; }
    red[t] = lsum; __syncthreads();
    for (int o = 128; o > 0; o >>= 1) { if (t < o) red[t] += red[t + o]; __syncthreads(); }
    float inv = 1.f / red[0];
    for (int c = t; c < 1000; c += 256) out[(size_t)b * 1000 + c] = logits[c] * inv;
}

// ============ learn_ac tail: [128,512]@[512,2] -> mean -> sigmoid -> coeff[b] ============
__global__ __launch_bounds__(128)
void ac_coeff_kernel(const u16* __restrict__ a, const float* __restrict__ W,
                     const float* __restrict__ bias, const float* __restrict__ ceps,
                     float* __restrict__ coeff)
{
    __shared__ float s0[128], s1[128];
    __shared__ float ac[2];
    int t = threadIdx.x;
    float d0 = 0.f, d1 = 0.f;
    for (int k = 0; k < 512; ++k) {
        float v = bf2f(a[(size_t)t * 512 + k]);
        d0 += v * W[k * 2]; d1 += v * W[k * 2 + 1];
    }
    s0[t] = d0; s1[t] = d1; __syncthreads();
    for (int o = 64; o > 0; o >>= 1) { if (t < o) { s0[t] += s0[t + o]; s1[t] += s1[t + o]; } __syncthreads(); }
    if (t == 0) {
        float m0 = s0[0] * (1.f / 128.f) + bias[0];
        float m1 = s1[0] * (1.f / 128.f) + bias[1];
        ac[0] = 1.f / (1.f + expf(-m0));
        ac[1] = 1.f / (1.f + expf(-m1));
    }
    __syncthreads();
    coeff[t] = ac[0] + ac[1] * ceps[t];
}

// ============ fp32 -> bf16 convert ============
__global__ void f32_to_bf16_kernel(const float* __restrict__ src, u16* __restrict__ dst, int n)
{
    int i = blockIdx.x * 256 + threadIdx.x;
    if (i < n) dst[i] = f2bf(src[i]);
}

// =====================================================================================
extern "C" void kernel_launch(void* const* d_in, const int* in_sizes, int n_in,
                              void* d_out, int out_size, void* d_ws, size_t ws_size,
                              hipStream_t stream)
{
    (void)in_sizes; (void)n_in; (void)out_size; (void)ws_size;
    const int N = 128, D = 512, M = N * D;  // M = 65536 rows for mixer GEMMs

    // ---- input indices (depth-first, insertion order; each Linear = w then b) ----
    const float* X   = (const float*)d_in[0];
    const float* wD0 = (const float*)d_in[1];  const float* bD0 = (const float*)d_in[2];
    const float* wD1 = (const float*)d_in[3];  const float* bD1 = (const float*)d_in[4];
    const float* wD2 = (const float*)d_in[5];  const float* bD2 = (const float*)d_in[6];
    const float* wA0 = (const float*)d_in[7];  const float* bA0 = (const float*)d_in[8];
    const float* wA1 = (const float*)d_in[9];  const float* bA1 = (const float*)d_in[10];
    const float* wA2 = (const float*)d_in[11]; const float* bA2 = (const float*)d_in[12];
    const float* wEm = (const float*)d_in[13]; const float* bEm = (const float*)d_in[14];
    // blocks: 8 sub-blocks at 15 + s*10: {ln.g, ln.b, m0.w, m0.b, m1.w, m1.b, f0.w, f0.b, f1.w, f1.b}
    const float* gF = (const float*)d_in[95];  const float* bF = (const float*)d_in[96];
    const float* wH = (const float*)d_in[97];  const float* bH = (const float*)d_in[98];
    const float* GE = (const float*)d_in[99];   // grid_eps [128,512,512]
    const float* CE = (const float*)d_in[100];  // coeff_eps [128]

    // ---- workspace carve-out ----
    char* cur = (char*)d_ws;
    auto take = [&](size_t bytes) { char* p = cur; cur += (bytes + 255) & ~(size_t)255; return p; };
    u16* xb    = (u16*)take((size_t)N * D * 2);
    u16* cwD0  = (u16*)take((size_t)D * D * 2);
    u16* cwD1  = (u16*)take((size_t)D * D * 2);
    u16* cwD2  = (u16*)take((size_t)D * 1024 * 2);
    u16* cwA0  = (u16*)take((size_t)D * D * 2);
    u16* cwA1  = (u16*)take((size_t)D * D * 2);
    u16* cwEm  = (u16*)take((size_t)D * D * 2);
    u16* cw_m0[8]; u16* cw_m1[8]; u16* cw_f0[8]; u16* cw_f1[8];
    for (int s = 0; s < 8; ++s) {
        int inner = (s & 1) ? 256 : 2048;
        cw_m0[s] = (u16*)take((size_t)D * D * 2);
        cw_m1[s] = (u16*)take((size_t)D * D * 2);
        cw_f0[s] = (u16*)take((size_t)D * inner * 2);
        cw_f1[s] = (u16*)take((size_t)inner * D * 2);
    }
    u16*   tA     = (u16*)take((size_t)N * D * 2);
    u16*   tB     = (u16*)take((size_t)N * D * 2);
    float* dbuf   = (float*)take((size_t)N * 1024 * 4);
    float* coeff  = (float*)take((size_t)N * 4);
    u16*   h      = (u16*)take((size_t)M * D * 2);
    u16*   xn     = (u16*)take((size_t)M * D * 2);
    u16*   t0     = (u16*)take((size_t)M * D * 2);
    u16*   mB     = (u16*)take((size_t)M * D * 2);
    u16*   f1     = (u16*)take((size_t)M * 2048 * 2);
    float* pooled = (float*)take((size_t)N * D * 4);

    auto conv = [&](const float* s, u16* d, size_t n) {
        f32_to_bf16_kernel<<<dim3((unsigned)((n + 255) / 256)), dim3(256), 0, stream>>>(s, d, (int)n);
    };
    // ---- convert inputs/weights to bf16 ----
    conv(X, xb, (size_t)N * D);
    conv(wD0, cwD0, (size_t)D * D); conv(wD1, cwD1, (size_t)D * D); conv(wD2, cwD2, (size_t)D * 1024);
    conv(wA0, cwA0, (size_t)D * D); conv(wA1, cwA1, (size_t)D * D); conv(wEm, cwEm, (size_t)D * D);
    for (int s = 0; s < 8; ++s) {
        int inner = (s & 1) ? 256 : 2048;
        conv((const float*)d_in[15 + s * 10 + 2], cw_m0[s], (size_t)D * D);
        conv((const float*)d_in[15 + s * 10 + 4], cw_m1[s], (size_t)D * D);
        conv((const float*)d_in[15 + s * 10 + 6], cw_f0[s], (size_t)D * inner);
        conv((const float*)d_in[15 + s * 10 + 8], cw_f1[s], (size_t)inner * D);
    }

    dim3 blk(256);
    #define GRID(MM, NN) dim3((NN) / BN, (MM) / BM)

    // ---- learn_D trunk: Linear->IF->ReLU x2, then Linear(512,1024)->sigmoid ----
    gemm_bf16_wmma<0, EP_IF><<<GRID(N, D), blk, 0, stream>>>(xb, cwD0, bD0, tA, N, D, D, nullptr, nullptr, nullptr, nullptr, nullptr);
    gemm_bf16_wmma<0, EP_IF><<<GRID(N, D), blk, 0, stream>>>(tA, cwD1, bD1, tB, N, D, D, nullptr, nullptr, nullptr, nullptr, nullptr);
    gemm_bf16_wmma<0, EP_SIG_F32><<<GRID(N, 1024), blk, 0, stream>>>(tB, cwD2, bD2, dbuf, N, 1024, D, nullptr, nullptr, nullptr, nullptr, nullptr);

    // ---- learn_ac trunk + coeff ----
    gemm_bf16_wmma<0, EP_IF><<<GRID(N, D), blk, 0, stream>>>(xb, cwA0, bA0, tA, N, D, D, nullptr, nullptr, nullptr, nullptr, nullptr);
    gemm_bf16_wmma<0, EP_IF><<<GRID(N, D), blk, 0, stream>>>(tA, cwA1, bA1, tB, N, D, D, nullptr, nullptr, nullptr, nullptr, nullptr);
    ac_coeff_kernel<<<dim3(1), dim3(128), 0, stream>>>(tB, wA2, bA2, CE, coeff);

    // ---- embed: h = grid @ W + b, grid generated on the fly (never materialized) ----
    gemm_bf16_wmma<1, EP_LIN><<<GRID(M, D), blk, 0, stream>>>(nullptr, cwEm, bEm, h, M, D, D, dbuf, coeff, GE, nullptr, nullptr);

    // ---- mixer blocks ----
    for (int s = 0; s < 8; ++s) {
        int inner = (s & 1) ? 256 : 2048;
        const float* ln_g = (const float*)d_in[15 + s * 10 + 0];
        const float* ln_b = (const float*)d_in[15 + s * 10 + 1];
        const float* m0b  = (const float*)d_in[15 + s * 10 + 3];
        const float* m1b  = (const float*)d_in[15 + s * 10 + 5];
        const float* f0b  = (const float*)d_in[15 + s * 10 + 7];
        const float* f1b  = (const float*)d_in[15 + s * 10 + 9];

        ln_kernel<<<dim3(M), blk, 0, stream>>>(h, ln_g, ln_b, xn);
        gemm_bf16_wmma<0, EP_IF_GELU><<<GRID(M, D), blk, 0, stream>>>(h, cw_m0[s], m0b, t0, M, D, D, nullptr, nullptr, nullptr, nullptr, nullptr);
        gemm_bf16_wmma<0, EP_IF_SIG><<<GRID(M, D), blk, 0, stream>>>(t0, cw_m1[s], m1b, mB, M, D, D, nullptr, nullptr, nullptr, nullptr, nullptr);
        gemm_bf16_wmma<0, EP_IF_GELU><<<GRID(M, inner), blk, 0, stream>>>(xn, cw_f0[s], f0b, f1, M, inner, D, nullptr, nullptr, nullptr, nullptr, nullptr);
        // fused: h = m * IF(f1 @ W + b) + h
        gemm_bf16_wmma<0, EP_RESID><<<GRID(M, D), blk, 0, stream>>>(f1, cw_f1[s], f1b, nullptr, M, D, inner, nullptr, nullptr, nullptr, mB, h);
    }

    // ---- final LN + token mean + head + softmax ----
    hipMemsetAsync(pooled, 0, (size_t)N * D * 4, stream);
    final_pool_kernel<<<dim3(M), blk, 0, stream>>>(h, gF, bF, pooled);
    head_softmax_kernel<<<dim3(N), blk, 0, stream>>>(pooled, wH, bH, (float*)d_out);
}
